// AUTOENCODER_MASK_MODF_SAGE_47382079209805
// MI455X (gfx1250) — matrix operand order repaired and checked by reading, not compiled, not run on hardware
//
#include <hip/hip_runtime.h>

// ---------------------------------------------------------------------------
// MI455X (gfx1250, wave32) masked-SAGE autoencoder.
// Edge phase: f32 global atomics into L2-resident ah[N,24].
// GEMM phase: bf16 x bf16 -> f32 via v_wmma_f32_16x16x32_bf16.
//   B (weight) panels staged into LDS once per workgroup via the Tensor Data
//   Mover (tensor_load_to_lds + s_wait_tensorcnt), 32KB K-slabs; A streamed
//   from global (L2-resident). LayerNorm+ReLU fused in-place; concat with
//   ah_norm fused by writing GEMM output into the next layer's padded input.
// ---------------------------------------------------------------------------

typedef __attribute__((ext_vector_type(16))) __bf16 v16bf;
typedef __attribute__((ext_vector_type(8)))  __bf16 v8bf;
typedef __attribute__((ext_vector_type(8)))  float  v8f;
typedef unsigned int uint4v __attribute__((ext_vector_type(4)));
typedef int          int8v  __attribute__((ext_vector_type(8)));
typedef int          int4v  __attribute__((ext_vector_type(4)));

#define LN_EPS 1e-5f
#define KC 256                      // K-slab staged in LDS (64 x 256 bf16 = 32KB)

// ---------------------------------------------------------------------------
__global__ void zero_f32(float* __restrict__ p, size_t n) {
    size_t i = (size_t)blockIdx.x * blockDim.x + threadIdx.x;
    if (i < n) p[i] = 0.0f;
}

// Per-edge scatter: msg = [x[src](10), dist, angle, feat(8), disc(4)] -> ah[dst]
__global__ void edge_scatter(const float* __restrict__ x,
                             const float* __restrict__ dist,
                             const float* __restrict__ ang,
                             const float* __restrict__ feat,
                             const float* __restrict__ disc,
                             const int* __restrict__ src,
                             const int* __restrict__ dst,
                             float* __restrict__ ah, int E) {
    int e = blockIdx.x * blockDim.x + threadIdx.x;
    if (e >= E) return;
    int s = src[e], d = dst[e];
    float* a = ah + (size_t)d * 24;
    const float* xs = x + (size_t)s * 10;
#pragma unroll
    for (int c = 0; c < 10; ++c) atomicAdd(a + c, xs[c]);
    atomicAdd(a + 10, dist[e]);
    atomicAdd(a + 11, ang[e]);
    const float* f = feat + (size_t)e * 8;
#pragma unroll
    for (int c = 0; c < 8; ++c) atomicAdd(a + 12 + c, f[c]);
    const float* q = disc + (size_t)e * 4;
#pragma unroll
    for (int c = 0; c < 4; ++c) atomicAdd(a + 20 + c, q[c]);
}

// Build bf16 concat inputs: in0 = [x | ahn | 0], and pre-fill ahn+pad columns
// of the enc1/dec1 buffer (cols 256..287) and the dec0 buffer (cols 512..543).
__global__ void node_prep(const float* __restrict__ x,
                          const float* __restrict__ ahf,
                          const float* __restrict__ nrm,
                          __bf16* __restrict__ in0,   // [N,64]
                          __bf16* __restrict__ in1,   // [N,288]
                          __bf16* __restrict__ ind0,  // [N,544]
                          int N) {
    int i = blockIdx.x * blockDim.x + threadIdx.x;
    if (i >= N) return;
    float s = nrm[i];
    __bf16* r0 = in0  + (size_t)i * 64;
    __bf16* r1 = in1  + (size_t)i * 288;
    __bf16* r2 = ind0 + (size_t)i * 544;
    const float* xi = x + (size_t)i * 10;
#pragma unroll
    for (int c = 0; c < 10; ++c) r0[c] = (__bf16)xi[c];
    const float* ai = ahf + (size_t)i * 24;
#pragma unroll
    for (int c = 0; c < 24; ++c) {
        __bf16 v = (__bf16)(ai[c] * s);
        r0[10 + c]  = v;
        r1[256 + c] = v;
        r2[512 + c] = v;
    }
#pragma unroll
    for (int c = 34; c < 64; ++c) r0[c] = (__bf16)0.0f;
#pragma unroll
    for (int c = 280; c < 288; ++c) r1[c] = (__bf16)0.0f;
#pragma unroll
    for (int c = 536; c < 544; ++c) r2[c] = (__bf16)0.0f;
}

__global__ void mask_tokens(const int* __restrict__ mask,
                            const float* __restrict__ token,
                            __bf16* __restrict__ in0, int M) {
    int j = blockIdx.x * blockDim.x + threadIdx.x;
    if (j >= M) return;
    __bf16* r = in0 + (size_t)mask[j] * 64;
#pragma unroll
    for (int c = 0; c < 10; ++c) r[c] = (__bf16)token[c];
}

// f32 [out,K] weight -> zero-padded bf16 [outPad,Kpad]; bias -> padded f32.
__global__ void pad_convert_w(const float* __restrict__ W,
                              const float* __restrict__ bias,
                              __bf16* __restrict__ Wd,
                              float* __restrict__ bd,
                              int out, int K, int outPad, int Kpad) {
    int idx = blockIdx.x * blockDim.x + threadIdx.x;
    int total = outPad * Kpad;
    if (idx < total) {
        int o = idx / Kpad, k = idx % Kpad;
        float v = (o < out && k < K) ? W[(size_t)o * K + k] : 0.0f;
        Wd[idx] = (__bf16)v;
    }
    if (bd && idx < outPad) bd[idx] = (bias && idx < out) ? bias[idx] : 0.0f;
}

// ---------------------------------------------------------------------------
// TDM: DMA `rows` rows of `kc` bf16 elements (global row stride K elements)
// into LDS at byte offset lds_off, packed with row stride kc. Descriptor per
// CDNA5 ISA 8.3/8.4: count=1, data_size=1 (2B), type=2, 2-D tile, groups 2/3
// zero. Issued once per wave; completion via TENSORcnt.
// This toolchain exposes the 6-arg builtin:
//   (uint32x4 g0, int32x8 g1, int32x4 g2, int32x4 g3, int32x8 g4, i32 cpol)
// ---------------------------------------------------------------------------
#if __has_builtin(__builtin_amdgcn_tensor_load_to_lds)
#define HAVE_TDM 1
__device__ __forceinline__ void tdm_load_panel(const __bf16* gsrc, unsigned lds_off,
                                               int kc, int rows, int K) {
    unsigned long long ga = (unsigned long long)(uintptr_t)gsrc;
    uint4v g0;
    g0[0] = 1u;                                              // count=1 (valid)
    g0[1] = lds_off;                                         // lds_addr
    g0[2] = (unsigned)(ga & 0xffffffffu);                    // global_addr[31:0]
    g0[3] = (unsigned)((ga >> 32) & 0x01ffffffu) | (2u << 30); // addr[56:32], type=2
    int8v g1;
    g1[0] = 1 << 16;                                         // data_size=1 (2 bytes)
    g1[1] = (int)(((unsigned)kc & 0xffffu) << 16);           // tensor_dim0[15:0]
    g1[2] = (int)((((unsigned)kc >> 16) & 0xffffu) |
                  (((unsigned)rows & 0xffffu) << 16));       // dim0 hi | tensor_dim1 lo
    g1[3] = (int)((((unsigned)rows >> 16) & 0xffffu) |
                  (((unsigned)kc & 0xffffu) << 16));         // dim1 hi | tile_dim0
    g1[4] = rows & 0xffff;                                   // tile_dim1 (tile_dim2=0)
    g1[5] = K;                                               // tensor_dim0_stride[31:0]
    g1[6] = 0;                                               // stride hi | dim1_stride lo
    g1[7] = 0;
    int4v z4 = {0, 0, 0, 0};
    int8v z8 = {0, 0, 0, 0, 0, 0, 0, 0};
    __builtin_amdgcn_tensor_load_to_lds(g0, g1, z4, z4, z8, 0);
}
#else
#define HAVE_TDM 0
#endif

// ---------------------------------------------------------------------------
// D[:, n0:n0+16*NT] = A[N,K] @ B[outPad,K]^T + bias, bf16 output at ldD.
// One wave = 16 rows x 16*NT cols; B panel staged in LDS in KC-wide slabs.
// A/B k order contiguous (k = 16*(lane/16) + s, same in both, not ISA table order); D f32 16x16 per ISA 7.12.2.
// ---------------------------------------------------------------------------
template <int NT>
__global__ void gemm_bf16(const __bf16* __restrict__ A,
                          const __bf16* __restrict__ B,
                          const float* __restrict__ bias,
                          __bf16* __restrict__ D,
                          int N, int K, int ldD) {
    __shared__ __align__(64) __bf16 Blds[4 * 16 * KC];   // 32KB max

    const int lane = threadIdx.x & 31;
    const int wave = threadIdx.x >> 5;
    const int m0 = (blockIdx.y * (blockDim.x >> 5) + wave) * 16;
    const bool active = (m0 < N);                 // wave-uniform
    const int n0 = blockIdx.x * (16 * NT);
    const int r  = lane & 15;
    const int hi = lane >> 4;

    const __bf16* aptr = A + (size_t)(m0 + r) * K + hi * 16;

    v8f acc[NT];
#pragma unroll
    for (int t = 0; t < NT; ++t) {
        v8f z = {0.f, 0.f, 0.f, 0.f, 0.f, 0.f, 0.f, 0.f};
        acc[t] = z;
    }

    const int rows = 16 * NT;
    for (int k0 = 0; k0 < K; k0 += KC) {
        const int kc = (K - k0 < KC) ? (K - k0) : KC;
        __syncthreads();                           // previous slab consumed
#if HAVE_TDM
        if (wave == 0) {                           // one TDM issue per block
            tdm_load_panel(B + (size_t)n0 * K + k0,
                           (unsigned)(uintptr_t)&Blds[0], kc, rows, K);
            __builtin_amdgcn_s_wait_tensorcnt(0);
        }
#else
        const int chunks = rows * (kc >> 3);       // 16B chunks
        for (int idx = threadIdx.x; idx < chunks; idx += blockDim.x) {
            int row = idx / (kc >> 3), c8 = idx % (kc >> 3);
            *(v8bf*)&Blds[row * kc + c8 * 8] =
                *(const v8bf*)(B + (size_t)(n0 + row) * K + k0 + c8 * 8);
        }
#endif
        __syncthreads();                           // slab visible

        if (active) {
            for (int k = 0; k < kc; k += 32) {
                v8bf alo = *(const v8bf*)(aptr + k0 + k);      // K k..k+7   (hi:+16)
                v8bf ahi = *(const v8bf*)(aptr + k0 + k + 8);  // K k+8..15  (hi:+16)
                v16bf a;
#pragma unroll
                for (int i = 0; i < 8; ++i) { a[i] = alo[i]; a[i + 8] = ahi[i]; }
#pragma unroll
                for (int t = 0; t < NT; ++t) {
                    // col j = t*16 + r, 16 contiguous K starting at k + hi*16
                    v16bf b = *(const v16bf*)&Blds[(t * 16 + r) * kc + k + hi * 16];
                    acc[t] = __builtin_amdgcn_wmma_f32_16x16x32_bf16(
                        false, a, false, b, (short)0, acc[t], false, false);
                }
            }
        }
    }

    if (active) {
#pragma unroll
        for (int t = 0; t < NT; ++t) {
            const int col = n0 + t * 16 + r;
            const float bb = bias ? bias[col] : 0.0f;
            __bf16* drow = D + (size_t)(m0 + hi * 8) * ldD + col;
#pragma unroll
            for (int i = 0; i < 8; ++i)
                drow[(size_t)i * (size_t)ldD] = (__bf16)(acc[t][i] + bb);
        }
    }
}

// In-place LayerNorm(+ReLU) over cols [0,n) of a bf16 row (ld >= n).
// One wave per row; n in {256,512}.
__global__ void ln_relu_wave(__bf16* __restrict__ Z, int ld, int n,
                             const float* __restrict__ g,
                             const float* __restrict__ beta,
                             int N, int relu) {
    int row = blockIdx.x * (blockDim.x >> 5) + (threadIdx.x >> 5);
    if (row >= N) return;
    int lane = threadIdx.x & 31;
    int per = n >> 5;                       // 8 or 16
    __bf16* z = Z + (size_t)row * ld + lane * per;
    float v[16];
    float s = 0.f;
    for (int i = 0; i < per; ++i) { v[i] = (float)z[i]; s += v[i]; }
    for (int off = 16; off; off >>= 1) s += __shfl_xor(s, off, 32);
    float mu = s / n;
    float q = 0.f;
    for (int i = 0; i < per; ++i) { float d = v[i] - mu; q += d * d; }
    for (int off = 16; off; off >>= 1) q += __shfl_xor(q, off, 32);
    float inv = rsqrtf(q / n + LN_EPS);
    int c0 = lane * per;
    for (int i = 0; i < per; ++i) {
        float y = (v[i] - mu) * inv * g[c0 + i] + beta[c0 + i];
        if (relu) y = fmaxf(y, 0.f);
        z[i] = (__bf16)y;
    }
}

// LayerNorm(+ReLU) for narrow rows (n <= 16): one thread per row, f32 output.
__global__ void ln_small(const __bf16* __restrict__ Z, int ldZ, int n,
                         const float* __restrict__ g,
                         const float* __restrict__ beta,
                         float* __restrict__ out, int ldO, int N, int relu) {
    int i = blockIdx.x * blockDim.x + threadIdx.x;
    if (i >= N) return;
    const __bf16* z = Z + (size_t)i * ldZ;
    float v[16], s = 0.f;
    for (int c = 0; c < n; ++c) { v[c] = (float)z[c]; s += v[c]; }
    float mu = s / n, q = 0.f;
    for (int c = 0; c < n; ++c) { float d = v[c] - mu; q += d * d; }
    float inv = rsqrtf(q / n + LN_EPS);
    float* o = out + (size_t)i * ldO;
    for (int c = 0; c < n; ++c) {
        float y = (v[c] - mu) * inv * g[c] + beta[c];
        if (relu) y = fmaxf(y, 0.f);
        o[c] = y;
    }
}

__global__ void gather_out(const int* __restrict__ mask,
                           const float* __restrict__ recon,
                           const float* __restrict__ x,
                           float* __restrict__ xp, float* __restrict__ xt, int M) {
    int j = blockIdx.x * blockDim.x + threadIdx.x;
    if (j >= M) return;
    int node = mask[j];
#pragma unroll
    for (int c = 0; c < 10; ++c) {
        xp[j * 10 + c] = recon[(size_t)node * 10 + c];
        xt[j * 10 + c] = x[(size_t)node * 10 + c];
    }
}

// ---------------------------------------------------------------------------
extern "C" void kernel_launch(void* const* d_in, const int* in_sizes, int n_in,
                              void* d_out, int out_size, void* d_ws, size_t ws_size,
                              hipStream_t stream) {
    (void)n_in; (void)out_size; (void)ws_size;
    const float* x        = (const float*)d_in[0];
    const float* distance = (const float*)d_in[1];
    const float* angle    = (const float*)d_in[2];
    const float* feat     = (const float*)d_in[3];
    const float* disc     = (const float*)d_in[4];
    const float* nrm      = (const float*)d_in[5];
    const int*   src      = (const int*)d_in[6];
    const int*   dst      = (const int*)d_in[7];
    const int*   mask     = (const int*)d_in[8];
    const float* token    = (const float*)d_in[9];
    const float* enc0_W = (const float*)d_in[10]; const float* enc0_b = (const float*)d_in[11];
    const float* enc1_W = (const float*)d_in[12]; const float* enc1_b = (const float*)d_in[13];
    const float* dec0_W = (const float*)d_in[14]; const float* dec0_b = (const float*)d_in[15];
    const float* dec1_W = (const float*)d_in[16]; const float* dec1_b = (const float*)d_in[17];
    const float* enc0_g = (const float*)d_in[18]; const float* enc0_be = (const float*)d_in[19];
    const float* enc1_g = (const float*)d_in[20]; const float* enc1_be = (const float*)d_in[21];
    const float* dec0_g = (const float*)d_in[22]; const float* dec0_be = (const float*)d_in[23];
    const float* dec1_g = (const float*)d_in[24]; const float* dec1_be = (const float*)d_in[25];
    const float* e2d_W  = (const float*)d_in[26];
    const float* np_W   = (const float*)d_in[27]; const float* np_b  = (const float*)d_in[28];
    const float* np_g   = (const float*)d_in[29]; const float* np_be = (const float*)d_in[30];

    const int N = in_sizes[5];   // 100000 (multiple of 16)
    const int E = in_sizes[1];   // 3200000
    const int M = in_sizes[8];   // 20000

    // ---- workspace carving (256B aligned) ----
    char* p = (char*)d_ws;
    auto take = [&](size_t bytes) { char* r = p; p += (bytes + 255) & ~(size_t)255; return r; };
    float*  ah   = (float*) take((size_t)N * 24 * 4);
    __bf16* in0  = (__bf16*)take((size_t)N * 64 * 2);    // enc0 input [x|ahn|0]
    __bf16* in1  = (__bf16*)take((size_t)N * 288 * 2);   // enc1 input; reused as dec1 input
    __bf16* h12  = (__bf16*)take((size_t)N * 512 * 2);   // enc1 out -> h (after LN+ReLU)
    __bf16* ind0 = (__bf16*)take((size_t)N * 544 * 2);   // dec0 input [rep|ahn|0]
    __bf16* znp  = (__bf16*)take((size_t)N * 16 * 2);
    __bf16* zd1  = (__bf16*)take((size_t)N * 16 * 2);
    float*  recon= (float*) take((size_t)N * 10 * 4);
    __bf16* w_enc0 = (__bf16*)take(256 * 64 * 2);  float* b_enc0 = (float*)take(256 * 4);
    __bf16* w_enc1 = (__bf16*)take(512 * 288 * 2); float* b_enc1 = (float*)take(512 * 4);
    __bf16* w_e2d  = (__bf16*)take(512 * 512 * 2);
    __bf16* w_np   = (__bf16*)take(16 * 512 * 2);  float* b_np   = (float*)take(16 * 4);
    __bf16* w_dec0 = (__bf16*)take(256 * 544 * 2); float* b_dec0 = (float*)take(256 * 4);
    __bf16* w_dec1 = (__bf16*)take(16 * 288 * 2);  float* b_dec1 = (float*)take(16 * 4);

    float* xp  = (float*)d_out;            // [M,10]
    float* xt  = xp + (size_t)M * 10;      // [M,10]
    float* nsc = xt + (size_t)M * 10;      // [N,16]

    // ---- weight prep (bf16, zero-padded K/out) ----
    auto padw = [&](const float* W, const float* b, __bf16* Wd, float* bd,
                    int out, int K, int outPad, int Kpad) {
        int total = outPad * Kpad;
        pad_convert_w<<<(total + 255) / 256, 256, 0, stream>>>(W, b, Wd, bd, out, K, outPad, Kpad);
    };
    padw(enc0_W, enc0_b, w_enc0, b_enc0, 256, 34, 256, 64);
    padw(enc1_W, enc1_b, w_enc1, b_enc1, 512, 280, 512, 288);
    padw(e2d_W, nullptr, w_e2d, nullptr, 512, 512, 512, 512);
    padw(np_W,  np_b,   w_np,   b_np,   16, 512, 16, 512);
    padw(dec0_W, dec0_b, w_dec0, b_dec0, 256, 536, 256, 544);
    padw(dec1_W, dec1_b, w_dec1, b_dec1, 10, 280, 16, 288);

    // ---- edge aggregation + node prep ----
    size_t nah = (size_t)N * 24;
    zero_f32<<<(int)((nah + 255) / 256), 256, 0, stream>>>(ah, nah);
    edge_scatter<<<(E + 255) / 256, 256, 0, stream>>>(x, distance, angle, feat, disc, src, dst, ah, E);
    node_prep<<<(N + 255) / 256, 256, 0, stream>>>(x, ah, nrm, in0, in1, ind0, N);
    mask_tokens<<<(M + 255) / 256, 256, 0, stream>>>(mask, token, in0, M);

    // ---- GEMM chain ----
    const int mtiles = (N + 15) / 16;          // 6250
    const int gy = (mtiles + 3) / 4;           // 4 waves/block, 1 row-tile each
    dim3 blk(128);

    // enc0: [N,64] x [256,64]^T -> in1 cols 0..255 (ld 288)
    gemm_bf16<4><<<dim3(256 / 64, gy), blk, 0, stream>>>(in0, w_enc0, b_enc0, in1, N, 64, 288);
    ln_relu_wave<<<(N + 7) / 8, 256, 0, stream>>>(in1, 288, 256, enc0_g, enc0_be, N, 1);
    // enc1: [N,288] x [512,288]^T -> h12 (ld 512)
    gemm_bf16<4><<<dim3(512 / 64, gy), blk, 0, stream>>>(in1, w_enc1, b_enc1, h12, N, 288, 512);
    ln_relu_wave<<<(N + 7) / 8, 256, 0, stream>>>(h12, 512, 512, enc1_g, enc1_be, N, 1);
    // node scores: [N,512] x [16,512]^T -> znp; LN -> d_out
    gemm_bf16<1><<<dim3(1, gy), blk, 0, stream>>>(h12, w_np, b_np, znp, N, 512, 16);
    ln_small<<<(N + 255) / 256, 256, 0, stream>>>(znp, 16, 16, np_g, np_be, nsc, 16, N, 0);
    // rep = h @ e2d^T -> ind0 cols 0..511 (ld 544)
    gemm_bf16<4><<<dim3(512 / 64, gy), blk, 0, stream>>>(h12, w_e2d, nullptr, ind0, N, 512, 544);
    // dec0: [N,544] x [256,544]^T -> in1 cols 0..255 (ahn cols still valid)
    gemm_bf16<4><<<dim3(256 / 64, gy), blk, 0, stream>>>(ind0, w_dec0, b_dec0, in1, N, 544, 288);
    ln_relu_wave<<<(N + 7) / 8, 256, 0, stream>>>(in1, 288, 256, dec0_g, dec0_be, N, 1);
    // dec1: [N,288] x [16,288]^T (rows 10..15 zero) -> zd1; LN+ReLU -> recon f32
    gemm_bf16<1><<<dim3(1, gy), blk, 0, stream>>>(in1, w_dec1, b_dec1, zd1, N, 288, 16);
    ln_small<<<(N + 255) / 256, 256, 0, stream>>>(zd1, 16, 10, dec1_g, dec1_be, recon, 10, N, 1);

    // ---- gather masked outputs ----
    gather_out<<<(M + 255) / 256, 256, 0, stream>>>(mask, recon, x, xp, xt, M);
}
